// SelfAttentionNetwork_44032004718713
// MI455X (gfx1250) — compile-verified, matrix operand order checked by reading, not run on hardware
//
#include <hip/hip_runtime.h>
#include <math.h>

typedef _Float16 half4 __attribute__((ext_vector_type(4)));
typedef _Float16 half8 __attribute__((ext_vector_type(8)));
typedef _Float16 v16h  __attribute__((ext_vector_type(16)));
typedef float    v8f   __attribute__((ext_vector_type(8)));

static constexpr int D_IN  = 1024;  // INPUT_DIM
static constexpr int H_DIM = 1024;  // HIDDEN_DIM
static constexpr int NB    = 8;     // BATCH
static constexpr int SQ    = 2048;  // SEQ

// LDS tile: 128 rows x 32 halves per k-step, padded to 40 halves (80B) per row
// -> rows stay 16B aligned and 16-lane b128 reads are bank-conflict free.
static constexpr int LDSW = 40;

// ---------------------------------------------------------------------------
// WMMA 16x16x32 f16 fragment from an LDS tile (row-major, stride LDSW halves).
// ISA 7.12.2 16-bit A layout: lane l holds row (l&15); lanes>=16 use K+8;
// elements 0..7 -> K+base..K+base+7, elements 8..15 -> K+base+16..K+base+23.
// B fragments (from the N-major copy) use the symmetric layout.
// ---------------------------------------------------------------------------
__device__ __forceinline__ v16h lds_frag16(const _Float16* __restrict__ s,
                                           int rc, int lane) {
  const _Float16* p = s + (rc + (lane & 15)) * LDSW + ((lane >> 4) << 3);
  union { v16h v; half8 h[2]; } u;
  u.h[0] = *reinterpret_cast<const half8*>(p);
  u.h[1] = *reinterpret_cast<const half8*>(p + 16);
  return u.v;
}

// ---------------------------------------------------------------------------
// Generic f16 GEMM, f32 accumulate:  C = scale * (A @ Bt^T) (+ bias)
//   A  : [M, K] row-major f16, leading dim lda, batch stride sA (elements)
//   Bt : [N, K] row-major f16 (math-B stored N-major), ldb, stride sB
//   C  : OUT_MODE 0 -> f32 [M,N]; 1 -> f16 [M,N]; 2 -> f16 V-transposed
//        (M encodes b*SQ+s, store at [((m/SQ)*H_DIM + n)*SQ + m%SQ])
// Block: 256 threads = 8 waves; block tile 128x128; wave tile 64x32 (4x2 WMMA)
// K staged through LDS with double-buffered async copies (ASYNCcnt).
// All dims are multiples of the tiles -> no bounds checks, EXEC all-ones.
// ---------------------------------------------------------------------------
template<int OUT_MODE, bool HAS_BIAS>
__global__ __launch_bounds__(256)
void wmma_gemm_kernel(const _Float16* __restrict__ A,  int lda, long long sA,
                      const _Float16* __restrict__ Bt, int ldb, long long sB,
                      const float* __restrict__ bias,
                      void* __restrict__ C, int ldc, long long sC,
                      int Kdim, float scale)
{
  __shared__ __align__(16) _Float16 As[2][128 * LDSW];
  __shared__ __align__(16) _Float16 Bs[2][128 * LDSW];

  const int t    = threadIdx.x;
  const int lane = t & 31;
  const int wave = t >> 5;
  const int blkM = blockIdx.x * 128;
  const int blkN = blockIdx.y * 128;
  const int row0 = (wave & 1) * 64;   // wave tile origin within block tile
  const int col0 = (wave >> 1) * 32;
  const int b    = blockIdx.z;
  A  += (long long)b * sA;
  Bt += (long long)b * sB;

  // --- async copy assignment: 2 threads per row, 32B (2 x b128) each --------
  const int crow  = t >> 1;           // 0..127
  const int chalf = (t & 1) << 4;     // 0 or 16 halves (0/32 bytes)
  const _Float16* gA = A  + (size_t)(blkM + crow) * lda + chalf;
  const _Float16* gB = Bt + (size_t)(blkN + crow) * ldb + chalf;
  const unsigned lA[2] = {
      (unsigned)(unsigned long long)(const void*)&As[0][crow * LDSW + chalf],
      (unsigned)(unsigned long long)(const void*)&As[1][crow * LDSW + chalf] };
  const unsigned lB[2] = {
      (unsigned)(unsigned long long)(const void*)&Bs[0][crow * LDSW + chalf],
      (unsigned)(unsigned long long)(const void*)&Bs[1][crow * LDSW + chalf] };

  auto issue = [&](int kofs, int buf) {
    const unsigned long long ga = (unsigned long long)(const void*)(gA + kofs);
    const unsigned long long gb = (unsigned long long)(const void*)(gB + kofs);
    // INST_OFFSET applies to both the LDS and global address (ISA 08 §4.4).
    asm volatile(
        "global_load_async_to_lds_b128 %0, %2, off\n\t"
        "global_load_async_to_lds_b128 %0, %2, off offset:16\n\t"
        "global_load_async_to_lds_b128 %1, %3, off\n\t"
        "global_load_async_to_lds_b128 %1, %3, off offset:16"
        :
        : "v"(lA[buf]), "v"(lB[buf]), "v"(ga), "v"(gb)
        : "memory");
  };

  v8f acc[4][2];
  const v8f vzero = {};
#pragma unroll
  for (int mi = 0; mi < 4; ++mi)
#pragma unroll
    for (int ni = 0; ni < 2; ++ni)
      acc[mi][ni] = vzero;

  const int nk = Kdim >> 5;
  issue(0, 0);
  asm volatile("s_wait_asynccnt 0x0" ::: "memory");
  __syncthreads();

  for (int i = 0; i < nk; ++i) {
    const int buf = i & 1;
    if (i + 1 < nk) issue((i + 1) << 5, buf ^ 1);

    v16h af[4], bf[2];
#pragma unroll
    for (int mi = 0; mi < 4; ++mi)
      af[mi] = lds_frag16(&As[buf][0], row0 + mi * 16, lane);
#pragma unroll
    for (int ni = 0; ni < 2; ++ni)
      bf[ni] = lds_frag16(&Bs[buf][0], col0 + ni * 16, lane);
#pragma unroll
    for (int mi = 0; mi < 4; ++mi)
#pragma unroll
      for (int ni = 0; ni < 2; ++ni)
        acc[mi][ni] = __builtin_amdgcn_wmma_f32_16x16x32_f16(
            false, af[mi], false, bf[ni], (short)0, acc[mi][ni], false, false);

    // my next-buffer copies done + everyone finished reading buf before it
    // gets overwritten by the next iteration's copies.
    asm volatile("s_wait_asynccnt 0x0" ::: "memory");
    __syncthreads();
  }

  // C/D layout: lane (l&15) = column, lanes>=16 hold rows +8; VGPR r = row+r.
  const int nlo = lane & 15;
  const int rhi = (lane >> 4) << 3;
#pragma unroll
  for (int mi = 0; mi < 4; ++mi)
#pragma unroll
    for (int ni = 0; ni < 2; ++ni) {
      const int n = blkN + col0 + ni * 16 + nlo;
      const float bv = HAS_BIAS ? bias[n] : 0.0f;
#pragma unroll
      for (int r = 0; r < 8; ++r) {
        const int m = blkM + row0 + mi * 16 + rhi + r;
        const float v = acc[mi][ni][r] * scale + bv;
        if (OUT_MODE == 0) {
          reinterpret_cast<float*>(C)[(long long)b * sC + (long long)m * ldc + n] = v;
        } else if (OUT_MODE == 1) {
          reinterpret_cast<_Float16*>(C)[(long long)b * sC + (long long)m * ldc + n] =
              (_Float16)v;
        } else {  // V stored transposed: [B][H_DIM][SQ]
          const int bb = m / SQ, s = m % SQ;
          reinterpret_cast<_Float16*>(C)[((long long)bb * H_DIM + n) * SQ + s] =
              (_Float16)v;
        }
      }
    }
}

// ---------------------------------------------------------------------------
// f32 -> f16 bulk convert (4 elements / thread)
// ---------------------------------------------------------------------------
__global__ __launch_bounds__(256)
void cvt_f32_to_f16_kernel(const float* __restrict__ src, _Float16* __restrict__ dst,
                           long long n4) {
  const long long i = (long long)blockIdx.x * 256 + threadIdx.x;
  if (i >= n4) return;
  const float4 f = reinterpret_cast<const float4*>(src)[i];
  half4 h;
  h[0] = (_Float16)f.x; h[1] = (_Float16)f.y; h[2] = (_Float16)f.z; h[3] = (_Float16)f.w;
  reinterpret_cast<half4*>(dst)[i] = h;
}

// ---------------------------------------------------------------------------
// Transpose + convert: src f32 [rows, cols] -> dst f16 [cols, rows]
// ---------------------------------------------------------------------------
__global__ __launch_bounds__(256)
void transpose_f32_to_f16_kernel(const float* __restrict__ src, _Float16* __restrict__ dst,
                                 int rows, int cols) {
  const int c = blockIdx.x * 32 + (threadIdx.x & 31);
  const int r = blockIdx.y * 8 + (threadIdx.x >> 5);
  if (r < rows && c < cols)
    dst[(size_t)c * rows + r] = (_Float16)src[(size_t)r * cols + c];
}

// ---------------------------------------------------------------------------
// Row softmax: one 256-thread block per row of length 2048 (8 elems/thread).
// Reads f32 scores, writes f16 probabilities.
// ---------------------------------------------------------------------------
__global__ __launch_bounds__(256)
void softmax_row_kernel(const float* __restrict__ sc, _Float16* __restrict__ P, int n) {
  __shared__ float red[8];
  const long long row = blockIdx.x;
  const float* src = sc + row * (long long)n;
  _Float16* dst = P + row * (long long)n;
  const int t = threadIdx.x;

  float v[8];
  float mx = -3.402823466e38f;
#pragma unroll
  for (int i = 0; i < 8; ++i) { v[i] = src[t + 256 * i]; mx = fmaxf(mx, v[i]); }
#pragma unroll
  for (int o = 16; o > 0; o >>= 1) mx = fmaxf(mx, __shfl_xor(mx, o, 32));
  if ((t & 31) == 0) red[t >> 5] = mx;
  __syncthreads();
  float m = red[0];
#pragma unroll
  for (int i = 1; i < 8; ++i) m = fmaxf(m, red[i]);
  __syncthreads();

  float sum = 0.0f;
#pragma unroll
  for (int i = 0; i < 8; ++i) { v[i] = __expf(v[i] - m); sum += v[i]; }
#pragma unroll
  for (int o = 16; o > 0; o >>= 1) sum += __shfl_xor(sum, o, 32);
  if ((t & 31) == 0) red[t >> 5] = sum;
  __syncthreads();
  float s = 0.0f;
#pragma unroll
  for (int i = 0; i < 8; ++i) s += red[i];
  const float inv = 1.0f / s;
#pragma unroll
  for (int i = 0; i < 8; ++i) dst[t + 256 * i] = (_Float16)(v[i] * inv);
}

// ---------------------------------------------------------------------------
// Host orchestration
// ---------------------------------------------------------------------------
extern "C" void kernel_launch(void* const* d_in, const int* in_sizes, int n_in,
                              void* d_out, int out_size, void* d_ws, size_t ws_size,
                              hipStream_t stream) {
  (void)in_sizes; (void)n_in; (void)out_size; (void)ws_size;

  const float* x  = (const float*)d_in[0];
  const float* Wq = (const float*)d_in[1];
  const float* bq = (const float*)d_in[2];
  const float* Wk = (const float*)d_in[3];
  const float* bk = (const float*)d_in[4];
  const float* Wv = (const float*)d_in[5];
  const float* bv = (const float*)d_in[6];

  char* ws = (char*)d_ws;
  size_t off = 0;
  auto carve = [&](size_t bytes) -> char* {
    char* p = ws + off;
    off += (bytes + 255) & ~(size_t)255;
    return p;
  };

  _Float16* xh  = (_Float16*)carve((size_t)NB * SQ * D_IN * 2);
  _Float16* WqT = (_Float16*)carve((size_t)D_IN * H_DIM * 2);
  _Float16* WkT = (_Float16*)carve((size_t)D_IN * H_DIM * 2);
  _Float16* WvT = (_Float16*)carve((size_t)D_IN * H_DIM * 2);
  _Float16* Qh  = (_Float16*)carve((size_t)NB * SQ * H_DIM * 2);
  _Float16* Kh  = (_Float16*)carve((size_t)NB * SQ * H_DIM * 2);
  _Float16* Vt  = (_Float16*)carve((size_t)NB * H_DIM * SQ * 2);   // [B][H][S]
  float*    SC  = (float*)   carve((size_t)NB * SQ * SQ * 4);
  _Float16* P   = (_Float16*)carve((size_t)NB * SQ * SQ * 2);

  // 1. x -> f16
  {
    const long long n4 = (long long)NB * SQ * D_IN / 4;
    cvt_f32_to_f16_kernel<<<(unsigned)((n4 + 255) / 256), 256, 0, stream>>>(x, xh, n4);
  }

  // 2. Weight transpose-converts: W [D,H] f32 -> Wt [H,D] f16 (N-major WMMA B)
  {
    dim3 g(H_DIM / 32, D_IN / 8, 1);
    transpose_f32_to_f16_kernel<<<g, 256, 0, stream>>>(Wq, WqT, D_IN, H_DIM);
    transpose_f32_to_f16_kernel<<<g, 256, 0, stream>>>(Wk, WkT, D_IN, H_DIM);
    transpose_f32_to_f16_kernel<<<g, 256, 0, stream>>>(Wv, WvT, D_IN, H_DIM);
  }

  // 3. Q/K/V projections: [B*S, D] @ [D, H] + bias
  //    Q, K stored row-major f16; V stored transposed ([B][H][S]) for P@V.
  {
    dim3 g((NB * SQ) / 128, H_DIM / 128, 1);
    wmma_gemm_kernel<1, true><<<g, 256, 0, stream>>>(
        xh, D_IN, 0, WqT, D_IN, 0, bq, Qh, H_DIM, 0, D_IN, 1.0f);
    wmma_gemm_kernel<1, true><<<g, 256, 0, stream>>>(
        xh, D_IN, 0, WkT, D_IN, 0, bk, Kh, H_DIM, 0, D_IN, 1.0f);
    wmma_gemm_kernel<2, true><<<g, 256, 0, stream>>>(
        xh, D_IN, 0, WvT, D_IN, 0, bv, Vt, 0, 0, D_IN, 1.0f);
  }

  // 4. scores = (Q @ K^T) / sqrt(H)  -> f32 [B,S,S]  (lives in the 192MB L2)
  {
    dim3 g(SQ / 128, SQ / 128, NB);
    wmma_gemm_kernel<0, false><<<g, 256, 0, stream>>>(
        Qh, H_DIM, (long long)SQ * H_DIM,
        Kh, H_DIM, (long long)SQ * H_DIM,
        nullptr,
        SC, SQ, (long long)SQ * SQ,
        H_DIM, 1.0f / 32.0f /* 1/sqrt(1024) */);
  }

  // 5. softmax rows -> P (f16)
  softmax_row_kernel<<<NB * SQ, 256, 0, stream>>>(SC, P, SQ);

  // 6. out = P @ V  -> f32 [B,S,H]
  {
    dim3 g(SQ / 128, H_DIM / 128, NB);
    wmma_gemm_kernel<0, false><<<g, 256, 0, stream>>>(
        P,  SQ, (long long)SQ * SQ,
        Vt, SQ, (long long)H_DIM * SQ,
        nullptr,
        d_out, H_DIM, (long long)SQ * H_DIM,
        SQ, 1.0f);
  }
}
